// GraphAttentionNetwork_13855564497181
// MI455X (gfx1250) — matrix operand order repaired and checked by reading, not compiled, not run on hardware
//
#include <hip/hip_runtime.h>
#include <hip/hip_bf16.h>
#include <math.h>

// ---------------- constants ----------------
#define NP     16384    // nodes
#define CC     88       // channels
#define CP     96       // padded channels (6 x 16)
#define HID    176      // mlp hidden
#define HIDP   192      // padded hidden (12 x 16)
#define NGRAPH 32
#define NGN    512      // nodes per graph
#define NHEAD  4
#define HD     22
#define HDP    32       // head dim padded to one 32-wide k tile
#define EA_N   131072
#define EE_N   262144
#define EDIM   197
#define EEMB   8
#define EEMBP  32       // padded edge-embedding stride (one 32-wide k tile)
#define PE_RAW 18
#define PE_DIM 10
#define ECHUNK 65536
#define EKP    224      // edge-mlp padded K: 96 (se) + 96 (de) + 32 (emb)

typedef __attribute__((ext_vector_type(16))) _Float16 v16h;
typedef __attribute__((ext_vector_type(8)))  float    v8f;

// ---------------- WMMA fragment helpers (contiguous k order, shared by A and B) ----------------
// A (16x32 f16): lane holds row m=lane&15; lanes 0-15 K {0..15}, lanes 16-31 K {16..31} (this code's order, not the ISA table's).
// xr must point at (row base + ktile + (lane>>4)*16); reads window [0,16).
__device__ inline v16h a_from_window(const float* __restrict__ xr) {
  v16h a;
#pragma unroll
  for (int v = 0; v < 8; ++v) {
    const int k0 = 2 * v;
    a[2 * v]     = (_Float16)xr[k0];
    a[2 * v + 1] = (_Float16)xr[k0 + 1];
  }
  return a;
}

// same A pattern but from an f16 (LDS) row window
__device__ inline v16h a_from_window_h(const _Float16* __restrict__ xr) {
  v16h a;
#pragma unroll
  for (int v = 0; v < 8; ++v) {
    const int k0 = 2 * v;
    a[2 * v]     = xr[k0];
    a[2 * v + 1] = xr[k0 + 1];
  }
  return a;
}

__device__ inline v8f wmma_f16(v16h a, v16h b, v8f c) {
  return __builtin_amdgcn_wmma_f32_16x16x32_f16(false, a, false, b, (short)0, c, false, false);
}

// Cooperative async copy of the (transposed, padded, f16) weight panel into LDS.
// Uses gfx1250 GLOBAL_LOAD_ASYNC_TO_LDS_B128 (ASYNCcnt) then s_wait_asynccnt + barrier.
// Single __shared__ array per kernel -> LDS base offset 0.
__device__ inline void async_copy_weights(const _Float16* __restrict__ Wt, int nbytes) {
  for (int off = threadIdx.x * 16; off < nbytes; off += blockDim.x * 16) {
    const void* g = (const char*)Wt + off;
    asm volatile("global_load_async_to_lds_b128 %0, %1, off"
                 :: "v"(off), "v"(g) : "memory");
  }
  asm volatile("s_wait_asynccnt 0x0" ::: "memory");
  __syncthreads();
}

// ---------------- weight prep: W (Kact x Nact, f32) -> Wt (Npad x Kpad, f16, transposed+padded) ----
__global__ void prep_w(const float* __restrict__ W, int Kact, int Nact, int Kpad, int total,
                       _Float16* __restrict__ Wt) {
  int i = blockIdx.x * blockDim.x + threadIdx.x;
  if (i >= total) return;
  int n = i / Kpad, k = i - n * Kpad;
  float w = (k < Kact && n < Nact) ? W[(size_t)k * Nact + n] : 0.0f;
  Wt[i] = (_Float16)w;
}

// edge-mlp weight (184 x 88) remapped to segment-aligned K=224, transposed to (96 x 224)
__global__ void prep_w_edge(const float* __restrict__ W, _Float16* __restrict__ Wt) {
  int i = blockIdx.x * blockDim.x + threadIdx.x;
  if (i >= CP * EKP) return;
  int n = i / EKP, k = i - n * EKP;
  int kk = -1;
  if (k < CC) kk = k;                               // h[se] segment
  else if (k >= 96 && k < 96 + CC) kk = k - 8;      // h[de] segment -> rows 88..175
  else if (k >= 192 && k < 192 + EEMB) kk = k - 16; // emb segment  -> rows 176..183
  float w = (kk >= 0 && n < CC) ? W[(size_t)kk * CC + n] : 0.0f;
  Wt[i] = (_Float16)w;
}

__global__ void prep_bias(const float* __restrict__ b, int Nact, int Npad,
                          float* __restrict__ bp) {
  int i = blockIdx.x * blockDim.x + threadIdx.x;
  if (i < Npad) bp[i] = (i < Nact) ? b[i] : 0.0f;
}

// ---------------- GEMM: Y[M x NT*16] = op(X[M x KP] @ Wt^T + biasP (+res)) ----------------
// grid = M/64 blocks, 128 threads (4 waves); each wave owns a 16-row stripe across all NT col
// tiles. Weight panel staged once per block via async global->LDS copy; B fragments then come
// from LDS (ds_load_b128), A fragments from global (b128 + cvt_pk_f16_f32).
template <int NT, int KP>
__global__ void gemm_node_t(const float* __restrict__ X, int ldx,
                            const _Float16* __restrict__ Wt,
                            const float* __restrict__ biasP,
                            const float* __restrict__ res,
                            float* __restrict__ Y, int ldy, int relu) {
  __shared__ _Float16 wsh[NT * 16 * KP];
  async_copy_weights(Wt, NT * 16 * KP * 2);

  const int wave    = threadIdx.x >> 5;
  const int ln      = threadIdx.x & 31;
  const int lnn     = ln & 15;
  const int khalf   = ln >> 4;
  const int rowBase = (blockIdx.x * 4 + wave) * 16;
  v8f acc[NT] = {};
  const float* xrow = X + (size_t)(rowBase + lnn) * ldx + khalf * 16;
  const _Float16* wcol = wsh + (size_t)lnn * KP + khalf * 16;
#pragma unroll
  for (int kt = 0; kt < KP; kt += 32) {
    v16h a = a_from_window(xrow + kt);
    const _Float16* wp = wcol + kt;
#pragma unroll
    for (int t = 0; t < NT; ++t) {
      v16h b = *(const v16h*)(wp + (size_t)t * 16 * KP);
      acc[t] = wmma_f16(a, b, acc[t]);
    }
  }
  const int mOff = khalf * 8;
#pragma unroll
  for (int t = 0; t < NT; ++t) {
    const int n  = t * 16 + lnn;
    const float bi = biasP[n];
#pragma unroll
    for (int i = 0; i < 8; ++i) {
      const int m = rowBase + mOff + i;
      float y = acc[t][i] + bi;
      if (res) y += res[(size_t)m * ldy + n];
      if (relu) y = fmaxf(y, 0.0f);
      Y[(size_t)m * ldy + n] = y;
    }
  }
}

// ---------------- edge-gather GEMM for final MLP layer 1 (K = 224, segment aligned) ------------
__global__ void gemm_edge(const int* __restrict__ se, const int* __restrict__ de,
                          const float* __restrict__ h, const float* __restrict__ EembP,
                          const _Float16* __restrict__ Wt, const float* __restrict__ biasP,
                          float* __restrict__ Y, int edgeBase) {
  __shared__ _Float16 wsh[6 * 16 * EKP];
  async_copy_weights(Wt, 6 * 16 * EKP * 2);

  const int wave    = threadIdx.x >> 5;
  const int ln      = threadIdx.x & 31;
  const int lnn     = ln & 15;
  const int khalf   = ln >> 4;
  const int rowBase = (blockIdx.x * 4 + wave) * 16;
  const int m = edgeBase + rowBase + lnn;
  const int s = se[m], d = de[m];
  const float* seg0 = h + (size_t)s * CP;            // k 0..95
  const float* seg1 = h + (size_t)d * CP;            // k 96..191
  const float* seg2 = EembP + (size_t)m * EEMBP;     // k 192..223
  const _Float16* wcol = wsh + (size_t)lnn * EKP + khalf * 16;
  v8f acc[6] = {};
#pragma unroll
  for (int kt = 0; kt < EKP; kt += 32) {
    const float* win = (kt < 96) ? (seg0 + kt)
                     : (kt < 192) ? (seg1 + (kt - 96))
                                  : seg2;            // uniform (kt is unrolled constant)
    v16h a = a_from_window(win + khalf * 16);
    const _Float16* wp = wcol + kt;
#pragma unroll
    for (int t = 0; t < 6; ++t) {
      v16h b = *(const v16h*)(wp + (size_t)t * 16 * EKP);
      acc[t] = wmma_f16(a, b, acc[t]);
    }
  }
  const int mOff = khalf * 8;
#pragma unroll
  for (int t = 0; t < 6; ++t) {
    const int n = t * 16 + lnn;
    const float bi = biasP[n];
#pragma unroll
    for (int i = 0; i < 8; ++i) {
      const int mm = rowBase + mOff + i;
      Y[(size_t)mm * CP + n] = fmaxf(acc[t][i] + bi, 0.0f);
    }
  }
}

// ---------------- elementwise / utility kernels ----------------
__global__ void zero_k(float* __restrict__ y, int n) {
  int i = blockIdx.x * blockDim.x + threadIdx.x;
  if (i < n) y[i] = 0.0f;
}

__global__ void add_k(const float* __restrict__ a, const float* __restrict__ b,
                      float* __restrict__ y, int n) {
  int i = blockIdx.x * blockDim.x + threadIdx.x;
  if (i < n) y[i] = a[i] + b[i];
}

// per-column mean/var over nRows (one block per column)
__global__ void bn_stats(const float* __restrict__ X, int ld, int nRows,
                         float* __restrict__ mean, float* __restrict__ var) {
  __shared__ float s1[256], s2[256];
  const int c = blockIdx.x;
  float a = 0.0f, b = 0.0f;
  for (int r = threadIdx.x; r < nRows; r += 256) {
    float x = X[(size_t)r * ld + c];
    a += x; b += x * x;
  }
  s1[threadIdx.x] = a; s2[threadIdx.x] = b;
  __syncthreads();
  for (int s = 128; s > 0; s >>= 1) {
    if (threadIdx.x < s) { s1[threadIdx.x] += s1[threadIdx.x + s]; s2[threadIdx.x] += s2[threadIdx.x + s]; }
    __syncthreads();
  }
  if (threadIdx.x == 0) {
    float m = s1[0] / nRows;
    mean[c] = m;
    var[c]  = s2[0] / nRows - m * m;
  }
}

__global__ void bn_apply(const float* __restrict__ X, float* __restrict__ Y,
                         const float* __restrict__ mean, const float* __restrict__ var,
                         const float* __restrict__ g, const float* __restrict__ b, int n) {
  int i = blockIdx.x * blockDim.x + threadIdx.x;
  if (i >= n) return;
  int c = i % CP;
  float y = 0.0f;
  if (c < CC) {
    y = (X[i] - mean[c]) * rsqrtf(var[c] + 1e-5f) * g[c] + b[c];
  }
  Y[i] = y;
}

// edge attr (nE x 197) @ AW (197 x 8) -> emb with stride + zero padding
__global__ void edge_embed(const float* __restrict__ attr, const float* __restrict__ AW,
                           float* __restrict__ emb, int nE, int stride) {
  int e = blockIdx.x * blockDim.x + threadIdx.x;
  if (e >= nE) return;
  float acc[EEMB];
#pragma unroll
  for (int j = 0; j < EEMB; ++j) acc[j] = 0.0f;
  const float* ar = attr + (size_t)e * EDIM;
  for (int d = 0; d < EDIM; ++d) {
    float a = ar[d];
    const float* w = AW + d * EEMB;
#pragma unroll
    for (int j = 0; j < EEMB; ++j) acc[j] += a * w[j];
  }
  float* er = emb + (size_t)e * stride;
#pragma unroll
  for (int j = 0; j < EEMB; ++j) er[j] = acc[j];
  for (int j = EEMB; j < stride; ++j) er[j] = 0.0f;
}

// h = [x(0..77), bn(pe)@pe_w + pe_b, zeros(8)]
__global__ void build_h(const float* __restrict__ xd, float* __restrict__ h,
                        const float* __restrict__ pm, const float* __restrict__ pv,
                        const float* __restrict__ pg, const float* __restrict__ pb,
                        const float* __restrict__ pw, const float* __restrict__ pbias) {
  int node = blockIdx.x * blockDim.x + threadIdx.x;
  if (node >= NP) return;
  const float* x = xd + (size_t)node * CP;   // x_dense has 96 cols
  float* ho = h + (size_t)node * CP;
  for (int c = 0; c < CP - PE_RAW; ++c) ho[c] = x[c];
  float pe[PE_RAW];
#pragma unroll
  for (int j = 0; j < PE_RAW; ++j)
    pe[j] = (x[CP - PE_RAW + j] - pm[j]) * rsqrtf(pv[j] + 1e-5f) * pg[j] + pb[j];
#pragma unroll
  for (int i = 0; i < PE_DIM; ++i) {
    float a = pbias[i];
    for (int j = 0; j < PE_RAW; ++j) a += pe[j] * pw[j * PE_DIM + i];
    ho[(CP - PE_RAW) + i] = a;
  }
  for (int c = CC; c < CP; ++c) ho[c] = 0.0f;
}

// GINE message + scatter-add: aggr[dst] += relu(h[src] + A_emb@ew + eb)
__global__ void msg_aggr(const float* __restrict__ h, const float* __restrict__ Aemb,
                         const int* __restrict__ src, const int* __restrict__ dst,
                         const float* __restrict__ ew, const float* __restrict__ eb,
                         float* __restrict__ aggr, int nE) {
  int e = blockIdx.x * blockDim.x + threadIdx.x;
  if (e >= nE) return;
  float emb[EEMB];
#pragma unroll
  for (int j = 0; j < EEMB; ++j) emb[j] = Aemb[(size_t)e * EEMB + j];
  const int s = src[e], d = dst[e];
  const float* hs = h + (size_t)s * CP;
  __builtin_prefetch(hs, 0, 3);   // global_prefetch_b8
  for (int c = 0; c < CC; ++c) {
    float ea = eb[c];
#pragma unroll
    for (int j = 0; j < EEMB; ++j) ea += emb[j] * ew[j * CC + c];
    float m = hs[c] + ea;
    if (m > 0.0f) atomicAdd(&aggr[(size_t)d * CP + c], m);
  }
}

// ---------------- WMMA flash attention ----------------
// One block per (graph, head); 8 waves x 16 query rows x 4 q-blocks.
// LDS: K f16 [512][32] (HD padded->32, zeros), V^T f16 [32][512], P bounce [8 waves][16][32].
// Per 32-key tile: 2 WMMAs (QK^T), cross-lane row max/sum over the 16-lane half-groups
// (D-layout rows live on fixed half-groups), online rescale, P -> LDS -> A-layout, 2 WMMAs (PV).
__global__ void attention(const float* __restrict__ q, const float* __restrict__ k,
                          const float* __restrict__ v, float* __restrict__ o) {
  extern __shared__ _Float16 sh[];
  _Float16* Ksh = sh;                 // 512*32
  _Float16* Vt  = sh + NGN * HDP;     // 32*512
  _Float16* Psh = Vt + NGN * HDP;     // 8*16*32
  const int g    = blockIdx.x >> 2;
  const int hh   = blockIdx.x & 3;
  const int base = g * NGN;
  const int co   = hh * HD;

  // stage K (padded) and V (transposed, padded) as f16
  for (int idx = threadIdx.x; idx < NGN * HDP; idx += blockDim.x) {
    const int r = idx >> 5, d = idx & 31;
    const float kv = (d < HD) ? k[(size_t)(base + r) * CP + co + d] : 0.0f;
    const float vv = (d < HD) ? v[(size_t)(base + r) * CP + co + d] : 0.0f;
    Ksh[r * HDP + d] = (_Float16)kv;
    Vt[d * NGN + r]  = (_Float16)vv;
  }
  __syncthreads();

  const int wave  = threadIdx.x >> 5;
  const int ln    = threadIdx.x & 31;
  const int lnn   = ln & 15;
  const int grp   = ln >> 4;
  const float scale = 0.2132007163556104f;  // 1/sqrt(22)
  _Float16* Pw = Psh + wave * 16 * 32;      // this wave's P tile

  for (int qb = wave; qb < NGN / 16; qb += 8) {
    const int qBase = qb * 16;
    // Q A-fragment (row = qBase+lnn). Channels >= HD read row-local garbage but multiply
    // zeroed K/V padding -> contribute 0.
    v16h qa = a_from_window(q + (size_t)(base + qBase + lnn) * CP + co + grp * 16);
    float mrow[8], lrow[8];
    v8f o0 = {}, o1 = {};
#pragma unroll
    for (int i = 0; i < 8; ++i) { mrow[i] = -1e30f; lrow[i] = 0.0f; }

    for (int kt = 0; kt < NGN / 32; ++kt) {
      // QK^T: two 16-key column tiles
      const int j0 = kt * 32 + lnn;
      v16h bk0 = *(const v16h*)(Ksh + (size_t)j0 * HDP + grp * 16);
      v16h bk1 = *(const v16h*)(Ksh + (size_t)(j0 + 16) * HDP + grp * 16);
      v8f s0 = wmma_f16(qa, bk0, (v8f){});
      v8f s1 = wmma_f16(qa, bk1, (v8f){});
      // online softmax per row (row m = i + 8*grp lives on one 16-lane half-group)
#pragma unroll
      for (int i = 0; i < 8; ++i) {
        float a0 = s0[i] * scale, a1 = s1[i] * scale;
        float t = fmaxf(a0, a1);
        t = fmaxf(t, __shfl_xor(t, 1, 32));
        t = fmaxf(t, __shfl_xor(t, 2, 32));
        t = fmaxf(t, __shfl_xor(t, 4, 32));
        t = fmaxf(t, __shfl_xor(t, 8, 32));
        const float mn   = fmaxf(mrow[i], t);
        const float corr = __expf(mrow[i] - mn);
        const float p0 = __expf(a0 - mn);
        const float p1 = __expf(a1 - mn);
        float rs = p0 + p1;
        rs += __shfl_xor(rs, 1, 32);
        rs += __shfl_xor(rs, 2, 32);
        rs += __shfl_xor(rs, 4, 32);
        rs += __shfl_xor(rs, 8, 32);
        lrow[i] = lrow[i] * corr + rs;
        mrow[i] = mn;
        o0[i] *= corr; o1[i] *= corr;
        const int m = i + 8 * grp;
        Pw[m * 32 + lnn]      = (_Float16)p0;
        Pw[m * 32 + 16 + lnn] = (_Float16)p1;
      }
      // P back as A-fragment (wave-private LDS; DS ops are in-order within a wave)
      v16h pa = a_from_window_h(Pw + lnn * 32 + grp * 16);
      // PV: V^T rows are output channels -> contiguous B fragments
      v16h bv0 = *(const v16h*)(Vt + (size_t)lnn * NGN + kt * 32 + grp * 16);
      v16h bv1 = *(const v16h*)(Vt + (size_t)(16 + lnn) * NGN + kt * 32 + grp * 16);
      o0 = wmma_f16(pa, bv0, o0);
      o1 = wmma_f16(pa, bv1, o1);
    }
    // epilogue: normalize and store (d = lnn and 16+lnn; only d < 22 are real)
#pragma unroll
    for (int i = 0; i < 8; ++i) {
      const float inv = 1.0f / lrow[i];
      const size_t row = (size_t)(base + qBase + i + 8 * grp);
      o[row * CP + co + lnn] = o0[i] * inv;
      if (lnn < HD - 16) o[row * CP + co + 16 + lnn] = o1[i] * inv;
    }
  }
}

// final logits: hid (chunk x 96) @ emlp_w2 (88 x 1) + b
__global__ void edge_logits(const float* __restrict__ hid, const float* __restrict__ w2,
                            const float* __restrict__ b2, float* __restrict__ out, int edgeBase) {
  int e = blockIdx.x * blockDim.x + threadIdx.x;
  if (e >= ECHUNK) return;
  const float* r = hid + (size_t)e * CP;
  float a = b2[0];
  for (int c = 0; c < CC; ++c) a += r[c] * w2[c];
  out[edgeBase + e] = a;
}

// ---------------- host launcher ----------------
extern "C" void kernel_launch(void* const* d_in, const int* in_sizes, int n_in,
                              void* d_out, int out_size, void* d_ws, size_t ws_size,
                              hipStream_t stream) {
  (void)in_sizes; (void)n_in; (void)out_size; (void)ws_size;
  const float* x_dense = (const float*)d_in[0];
  const int*   A_ei    = (const int*)d_in[1];
  const float* A_ea    = (const float*)d_in[2];
  const int*   E_ei    = (const int*)d_in[3];
  const float* E_ea    = (const float*)d_in[4];
  const float* AW      = (const float*)d_in[6];
  const float* pe_g    = (const float*)d_in[7];
  const float* pe_bet  = (const float*)d_in[8];
  const float* pe_w    = (const float*)d_in[9];
  const float* pe_b    = (const float*)d_in[10];
  const float* g_ew    = (const float*)d_in[11];
  const float* g_eb    = (const float*)d_in[12];
  const float* g_w1    = (const float*)d_in[13];
  const float* g_b1    = (const float*)d_in[14];
  const float* g_w2    = (const float*)d_in[15];
  const float* g_b2    = (const float*)d_in[16];
  const float* wq = (const float*)d_in[17]; const float* bq = (const float*)d_in[18];
  const float* wk = (const float*)d_in[19]; const float* bk = (const float*)d_in[20];
  const float* wv = (const float*)d_in[21]; const float* bv = (const float*)d_in[22];
  const float* wo = (const float*)d_in[23]; const float* bo = (const float*)d_in[24];
  const float* n1g = (const float*)d_in[25]; const float* n1b = (const float*)d_in[26];
  const float* n2g = (const float*)d_in[27]; const float* n2b = (const float*)d_in[28];
  const float* n3g = (const float*)d_in[29]; const float* n3b = (const float*)d_in[30];
  const float* mw1 = (const float*)d_in[31]; const float* mb1 = (const float*)d_in[32];
  const float* mw2 = (const float*)d_in[33]; const float* mb2 = (const float*)d_in[34];
  const float* ew1 = (const float*)d_in[35]; const float* eb1 = (const float*)d_in[36];
  const float* ew2 = (const float*)d_in[37]; const float* eb2 = (const float*)d_in[38];
  float* out = (float*)d_out;

  // ---------------- workspace layout ----------------
  float* ws = (float*)d_ws;
  size_t o = 0;
  float* f_Aemb  = ws + o; o += (size_t)EA_N * EEMB;
  float* f_EembP = ws + o; o += (size_t)EE_N * EEMBP;
  const size_t NB = (size_t)NP * CP + 64;   // +64 slack: attention Q window may read past row end
  float* f_h    = ws + o; o += NB;
  float* f_aggr = ws + o; o += NB;
  float* f_t0   = ws + o; o += NB;
  float* f_t1   = ws + o; o += NB;
  float* f_t2   = ws + o; o += NB;
  float* f_hl   = ws + o; o += NB;
  float* f_ha   = ws + o; o += NB;
  float* f_qb   = ws + o; o += NB;
  float* f_kb   = ws + o; o += NB;
  float* f_vb   = ws + o; o += NB;
  float* f_ob   = ws + o; o += NB;
  float* f_mh   = ws + o; o += (size_t)NP * HIDP;
  float* f_eh   = ws + o; o += (size_t)ECHUNK * CP;
  float* st_m   = ws + o; o += 128;
  float* st_v   = ws + o; o += 128;
  float* f_bias = ws + o; o += 8192;      // padded biases: 896 floats per layer + edge
  _Float16* hws = (_Float16*)(ws + o);    // f16 weight region (32B aligned: o is multiple of 8)
  // per layer: 6 x 96*96 + 192*96 + 96*192 = 92160 halves; +edge 96*224
  const size_t WSQ = (size_t)CP * CP;                  // 9216
  const size_t WL  = 6 * WSQ + 2 * (size_t)CP * HIDP;  // 92160
  _Float16* wt_e = hws + 6 * WL;

  const int EB = 256;
  const int nEl = (int)((size_t)NP * CP);
  const int gEl = (nEl + EB - 1) / EB;
  const dim3 gN(NP / 64);
  const size_t attn_lds = (size_t)(NGN * HDP + HDP * NGN + 8 * 16 * 32) * sizeof(_Float16);

  // ---------------- weight/bias prep (all tiny; L2 resident afterwards) ----------------
  for (int l = 0; l < 6; ++l) {
    _Float16* wl = hws + (size_t)l * WL;
    const size_t sq = (size_t)CC * CC;
    prep_w<<<(CP * CP + 255) / 256, 256, 0, stream>>>(g_w1 + l * sq, CC, CC, CP, CP * CP, wl + 0 * WSQ);
    prep_w<<<(CP * CP + 255) / 256, 256, 0, stream>>>(g_w2 + l * sq, CC, CC, CP, CP * CP, wl + 1 * WSQ);
    prep_w<<<(CP * CP + 255) / 256, 256, 0, stream>>>(wq + l * sq, CC, CC, CP, CP * CP, wl + 2 * WSQ);
    prep_w<<<(CP * CP + 255) / 256, 256, 0, stream>>>(wk + l * sq, CC, CC, CP, CP * CP, wl + 3 * WSQ);
    prep_w<<<(CP * CP + 255) / 256, 256, 0, stream>>>(wv + l * sq, CC, CC, CP, CP * CP, wl + 4 * WSQ);
    prep_w<<<(CP * CP + 255) / 256, 256, 0, stream>>>(wo + l * sq, CC, CC, CP, CP * CP, wl + 5 * WSQ);
    prep_w<<<(HIDP * CP + 255) / 256, 256, 0, stream>>>(mw1 + (size_t)l * CC * HID, CC, HID, CP,
                                                        HIDP * CP, wl + 6 * WSQ);
    prep_w<<<(CP * HIDP + 255) / 256, 256, 0, stream>>>(mw2 + (size_t)l * HID * CC, HID, CC, HIDP,
                                                        CP * HIDP, wl + 6 * WSQ + (size_t)HIDP * CP);
    float* bl = f_bias + (size_t)l * 896;
    prep_bias<<<1, CP, 0, stream>>>(g_b1 + l * CC, CC, CP, bl + 0 * CP);
    prep_bias<<<1, CP, 0, stream>>>(g_b2 + l * CC, CC, CP, bl + 1 * CP);
    prep_bias<<<1, CP, 0, stream>>>(bq + l * CC, CC, CP, bl + 2 * CP);
    prep_bias<<<1, CP, 0, stream>>>(bk + l * CC, CC, CP, bl + 3 * CP);
    prep_bias<<<1, CP, 0, stream>>>(bv + l * CC, CC, CP, bl + 4 * CP);
    prep_bias<<<1, CP, 0, stream>>>(bo + l * CC, CC, CP, bl + 5 * CP);
    prep_bias<<<1, CP, 0, stream>>>(mb2 + l * CC, CC, CP, bl + 6 * CP);
    prep_bias<<<1, HIDP, 0, stream>>>(mb1 + l * HID, HID, HIDP, bl + 7 * CP);  // 192 wide
  }
  prep_w_edge<<<(CP * EKP + 255) / 256, 256, 0, stream>>>(ew1, wt_e);
  float* eb1P = f_bias + 6 * 896;
  prep_bias<<<1, CP, 0, stream>>>(eb1, CC, CP, eb1P);

  // ---------------- edge attr embeddings (single pass over the big arrays) ----------------
  edge_embed<<<(EA_N + EB - 1) / EB, EB, 0, stream>>>(A_ea, AW, f_Aemb, EA_N, EEMB);
  edge_embed<<<(EE_N + EB - 1) / EB, EB, 0, stream>>>(E_ea, AW, f_EembP, EE_N, EEMBP);

  // ---------------- input projection: bn(pe) @ pe_w, concat ----------------
  bn_stats<<<PE_RAW, 256, 0, stream>>>(x_dense + (CP - PE_RAW), CP, NP, st_m, st_v);
  build_h<<<NP / EB, EB, 0, stream>>>(x_dense, f_h, st_m, st_v, pe_g, pe_bet, pe_w, pe_b);

  for (int l = 0; l < 6; ++l) {
    _Float16* wl = hws + (size_t)l * WL;
    float* bl = f_bias + (size_t)l * 896;
    const float* ew_l = g_ew + (size_t)l * EEMB * CC;
    const float* eb_l = g_eb + (size_t)l * CC;

    // GINE conv
    zero_k<<<gEl, EB, 0, stream>>>(f_aggr, nEl);
    msg_aggr<<<(EA_N + EB - 1) / EB, EB, 0, stream>>>(f_h, f_Aemb, A_ei, A_ei + EA_N,
                                                      ew_l, eb_l, f_aggr, EA_N);
    add_k<<<gEl, EB, 0, stream>>>(f_h, f_aggr, f_t0, nEl);
    gemm_node_t<6, 96><<<gN, 128, 0, stream>>>(f_t0, CP, wl + 0 * WSQ, bl + 0 * CP, nullptr, f_t2, CP, 1);
    gemm_node_t<6, 96><<<gN, 128, 0, stream>>>(f_t2, CP, wl + 1 * WSQ, bl + 1 * CP, f_h, f_t1, CP, 0);
    bn_stats<<<CC, 256, 0, stream>>>(f_t1, CP, NP, st_m, st_v);
    bn_apply<<<gEl, EB, 0, stream>>>(f_t1, f_hl, st_m, st_v, n1g + l * CC, n1b + l * CC, nEl);

    // attention branch (from pre-conv h)
    gemm_node_t<6, 96><<<gN, 128, 0, stream>>>(f_h, CP, wl + 2 * WSQ, bl + 2 * CP, nullptr, f_qb, CP, 0);
    gemm_node_t<6, 96><<<gN, 128, 0, stream>>>(f_h, CP, wl + 3 * WSQ, bl + 3 * CP, nullptr, f_kb, CP, 0);
    gemm_node_t<6, 96><<<gN, 128, 0, stream>>>(f_h, CP, wl + 4 * WSQ, bl + 4 * CP, nullptr, f_vb, CP, 0);
    zero_k<<<gEl, EB, 0, stream>>>(f_ob, nEl);
    attention<<<NGRAPH * NHEAD, 256, attn_lds, stream>>>(f_qb, f_kb, f_vb, f_ob);
    gemm_node_t<6, 96><<<gN, 128, 0, stream>>>(f_ob, CP, wl + 5 * WSQ, bl + 5 * CP, f_h, f_t1, CP, 0);
    bn_stats<<<CC, 256, 0, stream>>>(f_t1, CP, NP, st_m, st_v);
    bn_apply<<<gEl, EB, 0, stream>>>(f_t1, f_ha, st_m, st_v, n2g + l * CC, n2b + l * CC, nEl);

    // combine + MLP + final norm
    add_k<<<gEl, EB, 0, stream>>>(f_hl, f_ha, f_t0, nEl);
    gemm_node_t<12, 96><<<gN, 128, 0, stream>>>(f_t0, CP, wl + 6 * WSQ, bl + 7 * CP, nullptr,
                                                f_mh, HIDP, 1);
    gemm_node_t<6, 192><<<gN, 128, 0, stream>>>(f_mh, HIDP, wl + 6 * WSQ + (size_t)HIDP * CP,
                                                bl + 6 * CP, f_t0, f_t1, CP, 0);
    bn_stats<<<CC, 256, 0, stream>>>(f_t1, CP, NP, st_m, st_v);
    bn_apply<<<gEl, EB, 0, stream>>>(f_t1, f_h, st_m, st_v, n3g + l * CC, n3b + l * CC, nEl);
  }

  // ---------------- final edge MLP in chunks (gather fused into WMMA A load) ----------------
  const dim3 gE(ECHUNK / 64);
  for (int c = 0; c < EE_N / ECHUNK; ++c) {
    const int base = c * ECHUNK;
    gemm_edge<<<gE, 128, 0, stream>>>(E_ei, E_ei + EE_N, f_h, f_EembP, wt_e, eb1P, f_eh, base);
    edge_logits<<<ECHUNK / EB, EB, 0, stream>>>(f_eh, ew2, eb2, out, base);
  }
}
